// ShiftedWindowMSA_26302379721286
// MI455X (gfx1250) — compile-verified, matrix operand order checked by reading, not run on hardware
//
#include <hip/hip_runtime.h>

typedef unsigned short u16;
typedef unsigned int   u32;
typedef __attribute__((ext_vector_type(16))) __bf16 v16bf;
typedef __attribute__((ext_vector_type(8)))  float  v8f;
typedef __attribute__((ext_vector_type(4)))  u32    u32x4;
typedef __attribute__((ext_vector_type(4)))  float  f32x4;

// ---- constants ----
#define BATCH   32
#define HWDIM   56
#define LTOK    3136          // 56*56
#define EMBED   192
#define HEADS   6
#define HD      32
#define WTOK    49            // tokens per window
#define MTOT    100352        // BATCH*LTOK, divisible by 64
#define QSCALE  0.17677669529663687f   // 1/sqrt(32)
#define NEGBIG  (-1.0e9f)

// qkv_ws strides for layout [B, H, win(64), comp(3), t(49), d(32)]
#define S_B   1806336   // 6*64*3*49*32
#define S_H   301056    // 64*3*49*32
#define S_WIN 4704      // 3*49*32
#define S_CMP 1568      // 49*32

__device__ __forceinline__ u16 f2bf(float f) {
    u32 u = __float_as_uint(f);
    u += 0x7FFFu + ((u >> 16) & 1u);      // round-to-nearest-even
    return (u16)(u >> 16);
}
__device__ __forceinline__ u32 pack2(float a, float b) {
    return (u32)f2bf(a) | ((u32)f2bf(b) << 16);
}

// Load one 16x32 bf16 fragment (A layout; B uses same layout on B^T storage).
// lane 0..15  -> row (row0+lane),    K = kofs + {0..7, 16..23}
// lane 16..31 -> row (row0+lane-16), K = kofs + {8..15, 24..31}
__device__ __forceinline__ v16bf load_frag(const u16* p, int row0, int stride, int kofs) {
    const int lane = threadIdx.x & 31;
    const int r  = row0 + (lane & 15);
    const int kb = kofs + ((lane >> 4) << 3);
    union { v16bf v; u32x4 q[2]; } u;
    u.q[0] = *(const u32x4*)(p + r * stride + kb);
    u.q[1] = *(const u32x4*)(p + r * stride + kb + 16);
    return u.v;
}

__device__ __forceinline__ v8f wmma_bf16(v16bf a, v16bf b, v8f c) {
    return __builtin_amdgcn_wmma_f32_16x16x32_bf16(false, a, false, b, (short)0, c, false, false);
}

// ---------------------------------------------------------------------------
// Kernel 0: one-shot weight convert+transpose fp32[K,N] -> bf16 wT[N,K]
// ---------------------------------------------------------------------------
__global__ __launch_bounds__(256) void conv_w(const float* __restrict__ w,
                                              u16* __restrict__ wT, int K, int N) {
    int i = blockIdx.x * 256 + threadIdx.x;
    if (i < N * K) {
        int n = i / K, k = i % K;
        wT[i] = f2bf(w[(size_t)k * N + n]);
    }
}

// ---------------------------------------------------------------------------
// Kernel 1: QKV projection (100352x192 @ 192x576) + bias + Q-scale, fused
// cyclic shift + window partition scatter into qkv_ws [B,H,64,3,49,32] bf16.
// ---------------------------------------------------------------------------
__global__ __launch_bounds__(256) void qkv_gemm(const float* __restrict__ x,
                                                const u16* __restrict__ wqkvT, // [576,192]
                                                const float* __restrict__ b_qkv,
                                                u16* __restrict__ qkv_ws) {
    __shared__ alignas(16) u16 xa[64 * 32];     // A tile: 64 rows x 32 k
    __shared__ alignas(16) u16 wt[192 * 32];    // B^T tile: 192 n-rows x 32 k

    const int tid  = threadIdx.x;
    const int lane = tid & 31;
    const int wv   = tid >> 5;
    const int mbase = blockIdx.x * 64;
    const int nbase = blockIdx.y * 192;

    const v8f vzero = {0.f,0.f,0.f,0.f,0.f,0.f,0.f,0.f};
    v8f acc[6];
    #pragma unroll
    for (int i = 0; i < 6; ++i) acc[i] = vzero;

    const int mt = wv >> 1;                 // this wave's M sub-tile
    const int ng = (wv & 1) * 6;            // this wave's N sub-tile group

    for (int kk = 0; kk < 192; kk += 32) {
        // stage A: fp32 -> bf16, 8 elems/thread (2x float4 -> 1x uint4)
        {
            int r = tid >> 2, c8 = (tid & 3) * 8;
            const f32x4* s = (const f32x4*)(x + (size_t)(mbase + r) * EMBED + kk + c8);
            f32x4 v0 = s[0], v1 = s[1];
            u32x4 o = { pack2(v0.x, v0.y), pack2(v0.z, v0.w),
                        pack2(v1.x, v1.y), pack2(v1.z, v1.w) };
            *(u32x4*)(xa + r * 32 + c8) = o;
        }
        // stage B^T: straight bf16 b128 copies (pre-converted weights)
        #pragma unroll
        for (int j = 0; j < 3; ++j) {
            int i = tid + j * 256;
            int r = i >> 2, c8 = (i & 3) * 8;
            *(u32x4*)(wt + r * 32 + c8) =
                *(const u32x4*)(wqkvT + (size_t)(nbase + r) * EMBED + kk + c8);
        }
        __syncthreads();
        v16bf a = load_frag(xa, mt * 16, 32, 0);
        #pragma unroll
        for (int ti = 0; ti < 6; ++ti) {
            v16bf b = load_frag(wt, (ng + ti) * 16, 32, 0);
            acc[ti] = wmma_bf16(a, b, acc[ti]);
        }
        __syncthreads();
    }

    const int hi = lane >> 4, lo = lane & 15;

    // token offsets (shared by all 6 tiles of this wave)
    size_t tokOff[8];
    #pragma unroll
    for (int r = 0; r < 8; ++r) {
        int m  = mbase + mt * 16 + r + 8 * hi;        // token index
        int bb = m / LTOK, l = m % LTOK;
        int y  = l / HWDIM, xx = l % HWDIM;
        int ys = y + 53;  if (ys >= 56) ys -= 56;     // roll -3
        int xs = xx + 53; if (xs >= 56) xs -= 56;
        int win = (ys / 7) * 8 + (xs / 7);
        int t   = (ys % 7) * 7 + (xs % 7);
        tokOff[r] = (size_t)bb * S_B + (size_t)win * S_WIN + (size_t)t * HD;
    }
    #pragma unroll
    for (int ti = 0; ti < 6; ++ti) {
        int c    = nbase + (ng + ti) * 16 + lo;       // global channel 0..575
        int comp = c % 3;                              // q/k/v (fastest axis)
        int e    = c / 3;
        int h    = e >> 5, d = e & 31;
        float bias  = b_qkv[c];
        float scale = (comp == 0) ? QSCALE : 1.0f;
        size_t chOff = (size_t)h * S_H + (size_t)comp * S_CMP + d;
        #pragma unroll
        for (int r = 0; r < 8; ++r)
            qkv_ws[tokOff[r] + chOff] = f2bf((acc[ti][r] + bias) * scale);
    }
}

// ---------------------------------------------------------------------------
// Kernel 2: per-window attention.  1 wave per (b, h, window) = 12288 waves.
// attn_ws layout: [B, L, 192] bf16 (window-merge + reverse roll fused).
// ---------------------------------------------------------------------------
__global__ __launch_bounds__(32) void attn_win(const u16* __restrict__ qkv_ws,
                                               const float* __restrict__ rel_bias,
                                               u16* __restrict__ attn_ws) {
    __shared__ alignas(16) u16  qs[64 * 32];    // Q rows (padded), pre-scaled
    __shared__ alignas(16) u16  ks[64 * 32];    // K rows (padded)
    __shared__ alignas(16) u16  vt[32 * 64];    // V^T : dim x token (padded)
    __shared__ alignas(16) u16  at[64 * 64];    // attn bf16
    __shared__ float            sc[64 * 64];    // scores fp32

    const int lane = threadIdx.x;
    const int wid  = blockIdx.x;
    const int b   = wid / (HEADS * 64);
    const int rem = wid % (HEADS * 64);
    const int h   = rem / 64;
    const int w   = rem % 64;
    const int wy  = w >> 3, wx = w & 7;

    // zero the padding that matters
    for (int i = WTOK * 32 + lane; i < 64 * 32; i += 32) { qs[i] = 0; ks[i] = 0; }
    for (int i = lane; i < 32 * 15; i += 32) vt[(i / 15) * 64 + 49 + (i % 15)] = 0;

    const u16* src = qkv_ws + ((size_t)((b * HEADS + h) * 64 + w) * 3) * S_CMP;
    // Q, K: straight b128 copies (1568 bf16 = 196 uint4 each)
    for (int i = lane; i < 196; i += 32) {
        ((u32x4*)qs)[i] = ((const u32x4*)src)[i];
        ((u32x4*)ks)[i] = ((const u32x4*)(src + S_CMP))[i];
    }
    // V: transpose into vt[d][t]
    for (int i = lane; i < 784; i += 32) {
        int t = i >> 4;
        int d = (i & 15) << 1;
        u32 vw = ((const u32*)(src + 2 * S_CMP))[i];
        vt[d * 64 + t]       = (u16)(vw & 0xFFFFu);
        vt[(d + 1) * 64 + t] = (u16)(vw >> 16);
    }
    __syncthreads();

    // scores = Qs @ K^T  (4x4 tiles of 16x16, K=32 in one WMMA)
    const int hi = lane >> 4, lo = lane & 15;
    #pragma unroll
    for (int mt = 0; mt < 4; ++mt) {
        v16bf a = load_frag(qs, mt * 16, 32, 0);
        #pragma unroll
        for (int nt = 0; nt < 4; ++nt) {
            v16bf bk = load_frag(ks, nt * 16, 32, 0);
            v8f c = {0.f,0.f,0.f,0.f,0.f,0.f,0.f,0.f};
            c = wmma_bf16(a, bk, c);
            #pragma unroll
            for (int r = 0; r < 8; ++r)
                sc[(mt * 16 + r + 8 * hi) * 64 + nt * 16 + lo] = c[r];
        }
    }
    __syncthreads();

    // softmax over 49 cols, + rel_bias + shift masks; write bf16 attn
    const bool rmask = (wy == 7), cmask = (wx == 7);
    for (int m = lane; m < WTOK; m += 32) {
        float* row = &sc[m * 64];
        const float* brow = rel_bias + m * WTOK;
        int m7 = m % 7;
        float mx = -3.0e38f;
        for (int n = 0; n < WTOK; ++n) {
            float s = row[n] + brow[n];
            if (rmask && ((m >= 28) != (n >= 28)))      s += NEGBIG;
            if (cmask && ((m7 >= 4) != ((n % 7) >= 4))) s += NEGBIG;
            row[n] = s;
            mx = fmaxf(mx, s);
        }
        float sum = 0.f;
        for (int n = 0; n < WTOK; ++n) { float e = __expf(row[n] - mx); row[n] = e; sum += e; }
        float inv = 1.0f / sum;
        for (int n = 0; n < WTOK; ++n) at[m * 64 + n] = f2bf(row[n] * inv);
        for (int n = WTOK; n < 64; ++n) at[m * 64 + n] = 0;   // K-dim padding
    }
    __syncthreads();

    // out = attn @ V  (4x2 tiles, K=64 in two WMMA steps); fused un-shift store
    #pragma unroll
    for (int mt = 0; mt < 4; ++mt) {
        size_t dstOff[8];
        bool   valid[8];
        #pragma unroll
        for (int r = 0; r < 8; ++r) {
            int m = mt * 16 + r + 8 * hi;
            valid[r] = (m < WTOK);
            int mm = valid[r] ? m : 0;
            int y  = wy * 7 + mm / 7 + 3; if (y  >= 56) y  -= 56;   // roll +3
            int xx = wx * 7 + mm % 7 + 3; if (xx >= 56) xx -= 56;
            dstOff[r] = ((size_t)b * LTOK + y * HWDIM + xx) * EMBED + h * HD;
        }
        #pragma unroll
        for (int nt = 0; nt < 2; ++nt) {
            v8f c = {0.f,0.f,0.f,0.f,0.f,0.f,0.f,0.f};
            #pragma unroll
            for (int kt = 0; kt < 2; ++kt) {
                v16bf a  = load_frag(at, mt * 16, 64, kt * 32);
                v16bf bv = load_frag(vt, nt * 16, 64, kt * 32);
                c = wmma_bf16(a, bv, c);
            }
            #pragma unroll
            for (int r = 0; r < 8; ++r)
                if (valid[r])
                    attn_ws[dstOff[r] + nt * 16 + lo] = f2bf(c[r]);
        }
    }
}

// ---------------------------------------------------------------------------
// Kernel 3: output projection (100352x192 @ 192x192) + bias -> fp32 d_out
// ---------------------------------------------------------------------------
__global__ __launch_bounds__(256) void out_gemm(const u16* __restrict__ attn_ws,
                                                const u16* __restrict__ woutT,  // [192,192]
                                                const float* __restrict__ b_out,
                                                float* __restrict__ out) {
    __shared__ alignas(16) u16 aa[64 * 32];
    __shared__ alignas(16) u16 wt[192 * 32];

    const int tid  = threadIdx.x;
    const int lane = tid & 31;
    const int wv   = tid >> 5;
    const int mbase = blockIdx.x * 64;

    const v8f vzero = {0.f,0.f,0.f,0.f,0.f,0.f,0.f,0.f};
    v8f acc[6];
    #pragma unroll
    for (int i = 0; i < 6; ++i) acc[i] = vzero;

    const int mt = wv >> 1;
    const int ng = (wv & 1) * 6;

    for (int kk = 0; kk < 192; kk += 32) {
        {   // stage A: bf16 b128 copies
            int r = tid >> 2, c8 = (tid & 3) * 8;
            *(u32x4*)(aa + r * 32 + c8) =
                *(const u32x4*)(attn_ws + (size_t)(mbase + r) * EMBED + kk + c8);
        }
        #pragma unroll
        for (int j = 0; j < 3; ++j) {   // stage B^T: bf16 b128 copies
            int i = tid + j * 256;
            int r = i >> 2, c8 = (i & 3) * 8;
            *(u32x4*)(wt + r * 32 + c8) =
                *(const u32x4*)(woutT + (size_t)(ng * 0 + r) * EMBED + kk + c8);
        }
        __syncthreads();
        v16bf a = load_frag(aa, mt * 16, 32, 0);
        #pragma unroll
        for (int ti = 0; ti < 6; ++ti) {
            v16bf b = load_frag(wt, (ng + ti) * 16, 32, 0);
            acc[ti] = wmma_bf16(a, b, acc[ti]);
        }
        __syncthreads();
    }

    const int hi = lane >> 4, lo = lane & 15;
    #pragma unroll
    for (int ti = 0; ti < 6; ++ti) {
        int n = (ng + ti) * 16 + lo;
        float bias = b_out[n];
        #pragma unroll
        for (int r = 0; r < 8; ++r) {
            int m = mbase + mt * 16 + r + 8 * hi;
            out[(size_t)m * EMBED + n] = acc[ti][r] + bias;
        }
    }
}

// ---------------------------------------------------------------------------
extern "C" void kernel_launch(void* const* d_in, const int* in_sizes, int n_in,
                              void* d_out, int out_size, void* d_ws, size_t ws_size,
                              hipStream_t stream) {
    const float* x        = (const float*)d_in[0];
    const float* w_qkv    = (const float*)d_in[1];
    const float* b_qkv    = (const float*)d_in[2];
    const float* w_out    = (const float*)d_in[3];
    const float* b_out    = (const float*)d_in[4];
    const float* rel_bias = (const float*)d_in[5];

    // workspace: qkv bf16 [B,H,64,3,49,32] | attn bf16 [B,L,192] | wqkvT | woutT
    u16* qkv_ws  = (u16*)d_ws;
    const size_t QKV_ELEMS  = (size_t)BATCH * HEADS * 64 * 3 * WTOK * HD;  // 57,802,752
    const size_t ATTN_ELEMS = (size_t)MTOT * EMBED;                        // 19,267,584
    u16* attn_ws = qkv_ws + QKV_ELEMS;
    u16* wqkvT   = attn_ws + ATTN_ELEMS;           // [576,192]
    u16* woutT   = wqkvT + (size_t)576 * EMBED;    // [192,192]
    float* out   = (float*)d_out;

    conv_w<<<(576 * 192 + 255) / 256, 256, 0, stream>>>(w_qkv, wqkvT, EMBED, 576);
    conv_w<<<(192 * 192 + 255) / 256, 256, 0, stream>>>(w_out, woutT, EMBED, EMBED);
    qkv_gemm<<<dim3(MTOT / 64, 3), 256, 0, stream>>>(x, wqkvT, b_qkv, qkv_ws);
    attn_win<<<BATCH * HEADS * 64, 32, 0, stream>>>(qkv_ws, rel_bias, attn_ws);
    out_gemm<<<MTOT / 64, 256, 0, stream>>>(attn_ws, woutT, b_out, out);
}
